// MOELayer_86715389706430
// MI455X (gfx1250) — compile-verified, matrix operand order checked by reading, not run on hardware
//
#include <hip/hip_runtime.h>

// ---------------------------------------------------------------------------
// MoE grouped two-layer GEMM for MI455X (gfx1250, wave32, WMMA).
//
// Roofline: 68.7 GFLOP, ~0.64 GB min HBM traffic -> ~27us floor @ 23.3 TB/s.
// fp32 WMMA is K=4 only (8x slower than bf16 K=32), so we use split-bf16
// (hi+lo, 3 products) on v_wmma_f32_16x16x32_bf16 for ~fp32-grade accuracy
// at 3x bf16 cost -- still ~2.7x faster than the exact-fp32 WMMA path.
// ---------------------------------------------------------------------------

typedef __attribute__((ext_vector_type(16))) __bf16 v16bf;
typedef __attribute__((ext_vector_type(8)))  __bf16 v8bf;
typedef __attribute__((ext_vector_type(4)))  __bf16 v4bf;
typedef __attribute__((ext_vector_type(8)))  float  v8f;

#define NUM_EXPERT 16
#define IN_FEAT    1024
#define HID_FEAT   4096
#define OUT_FEAT   1024
#define T_TOKENS   4096
#define CAP        (T_TOKENS / NUM_EXPERT)   // 256 tokens per expert

#define BM   128     // block tile M
#define BN   128     // block tile N
#define BK   32      // block tile K (fp32 elements)
#define LDSK 40      // padded LDS k-stride (80B row stride, 16B aligned)

// ---- permutation build: group tokens by expert (order within group is
// irrelevant: each token's output depends only on its own expert weights) ----
__global__ void zero_counts_kernel(int* __restrict__ cnt) {
    if (threadIdx.x < NUM_EXPERT) cnt[threadIdx.x] = 0;
}

__global__ void build_perm_kernel(const int* __restrict__ gate,
                                  int* __restrict__ cnt,
                                  int* __restrict__ perm) {
    int t = blockIdx.x * blockDim.x + threadIdx.x;
    if (t < T_TOKENS) {
        int e = gate[t];
        int slot = atomicAdd(&cnt[e], 1);
        perm[e * CAP + slot] = t;
    }
}

// load one 16x32 bf16 WMMA fragment row: lane owns row (m or n) = base+lane%16,
// K chunks {c0..c0+7} and {c0+16..c0+23}, c0 = lane<16 ? 0 : 8.
__device__ __forceinline__ v16bf frag_load(const __bf16* __restrict__ rowbase, int c0) {
    v8bf x0 = *(const v8bf*)(rowbase + c0);        // 16B aligned
    v8bf x1 = *(const v8bf*)(rowbase + c0 + 16);   // +32B
    return __builtin_shufflevector(x0, x1, 0, 1, 2, 3, 4, 5, 6, 7,
                                            8, 9, 10, 11, 12, 13, 14, 15);
}

// split a float4 into bf16 hi + bf16 residual lo, store 8B each into LDS
__device__ __forceinline__ void cvt_split4(float4 v,
                                           __bf16* __restrict__ hi,
                                           __bf16* __restrict__ lo) {
    float f0 = v.x, f1 = v.y, f2 = v.z, f3 = v.w;
    v4bf hv, lv;
    __bf16 h0 = (__bf16)f0, h1 = (__bf16)f1, h2 = (__bf16)f2, h3 = (__bf16)f3;
    hv[0] = h0; hv[1] = h1; hv[2] = h2; hv[3] = h3;
    lv[0] = (__bf16)(f0 - (float)h0);
    lv[1] = (__bf16)(f1 - (float)h1);
    lv[2] = (__bf16)(f2 - (float)h2);
    lv[3] = (__bf16)(f3 - (float)h3);
    *(v4bf*)hi = hv;   // 8B aligned ds_store
    *(v4bf*)lo = lv;
}

// C[e-group] = A * B^T, A is MxK row-major (gathered through perm if GATHER_A),
// B is NxK row-major (nn.Linear weights), C scattered through perm if SCATTER_C.
template <bool GATHER_A, bool SCATTER_C>
__global__ __launch_bounds__(256)
void moe_gemm_kernel(const float* __restrict__ A,
                     const float* __restrict__ Bw,
                     float* __restrict__ C,
                     const int* __restrict__ perm,
                     int N, int K) {
    __shared__ __bf16 sAhi[BM * LDSK];
    __shared__ __bf16 sAlo[BM * LDSK];
    __shared__ __bf16 sBhi[BN * LDSK];
    __shared__ __bf16 sBlo[BN * LDSK];

    const int tid   = threadIdx.x;
    const int e     = blockIdx.z;
    const int mBase = blockIdx.y * BM;
    const int nBase = blockIdx.x * BN;

    // ---- per-thread global->LDS slots: 4 float4 of A, 4 float4 of B -------
    const float* aPtr[4];
    const float* bPtr[4];
    int ldsOff[4];
#pragma unroll
    for (int i = 0; i < 4; ++i) {
        int linear = tid + i * 256;      // 1024 float4 per 128x32 tile
        int r  = linear >> 3;            // row 0..127
        int c4 = linear & 7;             // float4 column (of 8)
        int m = mBase + r;
        const float* arow;
        if (GATHER_A) arow = A + (size_t)perm[e * CAP + m] * (size_t)K;
        else          arow = A + (size_t)(e * CAP + m) * (size_t)K;
        aPtr[i] = arow + c4 * 4;
        bPtr[i] = Bw + ((size_t)e * (size_t)N + (size_t)(nBase + r)) * (size_t)K + c4 * 4;
        ldsOff[i] = r * LDSK + c4 * 4;
    }

    // ---- wave decomposition: 8 waves = 2(M) x 4(N); wave tile 64x32 -------
    const int lane   = tid & 31;
    const int wave   = tid >> 5;
    const int wm     = wave & 1;
    const int wn     = wave >> 1;
    const int l15    = lane & 15;
    const int c0     = (lane < 16) ? 0 : 8;

    v8f acc[4][2];
#pragma unroll
    for (int mt = 0; mt < 4; ++mt)
#pragma unroll
        for (int nt = 0; nt < 2; ++nt)
#pragma unroll
            for (int j = 0; j < 8; ++j) acc[mt][nt][j] = 0.0f;

    const int nk = K / BK;
    float4 aReg[4], bReg[4];
#pragma unroll
    for (int i = 0; i < 4; ++i) {
        aReg[i] = *(const float4*)(aPtr[i]);
        bReg[i] = *(const float4*)(bPtr[i]);
    }

    for (int kt = 0; kt < nk; ++kt) {
        // stage current k-tile into LDS as bf16 hi/lo
#pragma unroll
        for (int i = 0; i < 4; ++i) {
            cvt_split4(aReg[i], &sAhi[ldsOff[i]], &sAlo[ldsOff[i]]);
            cvt_split4(bReg[i], &sBhi[ldsOff[i]], &sBlo[ldsOff[i]]);
        }
        __syncthreads();

        // issue next tile's global loads before compute (overlap with WMMA)
        if (kt + 1 < nk) {
            size_t koff = (size_t)(kt + 1) * BK;
#pragma unroll
            for (int i = 0; i < 4; ++i) {
                aReg[i] = *(const float4*)(aPtr[i] + koff);
                bReg[i] = *(const float4*)(bPtr[i] + koff);
            }
        }

        // fragments
        v16bf ah[4], al[4], bh[2], bl[2];
#pragma unroll
        for (int mt = 0; mt < 4; ++mt) {
            int row = (wm * 64 + mt * 16 + l15) * LDSK;
            ah[mt] = frag_load(&sAhi[row], c0);
            al[mt] = frag_load(&sAlo[row], c0);
        }
#pragma unroll
        for (int nt = 0; nt < 2; ++nt) {
            int row = (wn * 32 + nt * 16 + l15) * LDSK;
            bh[nt] = frag_load(&sBhi[row], c0);
            bl[nt] = frag_load(&sBlo[row], c0);
        }

        // split-bf16 product: hi*hi + hi*lo + lo*hi  (24 WMMAs per k-step)
#pragma unroll
        for (int mt = 0; mt < 4; ++mt)
#pragma unroll
            for (int nt = 0; nt < 2; ++nt) {
                acc[mt][nt] = __builtin_amdgcn_wmma_f32_16x16x32_bf16(
                    false, ah[mt], false, bh[nt], (short)0, acc[mt][nt], false, false);
                acc[mt][nt] = __builtin_amdgcn_wmma_f32_16x16x32_bf16(
                    false, ah[mt], false, bl[nt], (short)0, acc[mt][nt], false, false);
                acc[mt][nt] = __builtin_amdgcn_wmma_f32_16x16x32_bf16(
                    false, al[mt], false, bh[nt], (short)0, acc[mt][nt], false, false);
            }
        __syncthreads();
    }

    // ---- epilogue: D layout -> VGPR j holds M=j (lanes 0-15) / M=j+8 ------
    const int lhalf8 = (lane >> 4) * 8;
#pragma unroll
    for (int mt = 0; mt < 4; ++mt) {
#pragma unroll
        for (int j = 0; j < 8; ++j) {
            int m = mBase + wm * 64 + mt * 16 + j + lhalf8;
            size_t crow;
            if (SCATTER_C) crow = (size_t)perm[e * CAP + m] * (size_t)N;
            else           crow = (size_t)(e * CAP + m) * (size_t)N;
#pragma unroll
            for (int nt = 0; nt < 2; ++nt) {
                int n = nBase + wn * 32 + nt * 16 + l15;
                C[crow + (size_t)n] = acc[mt][nt][j];
            }
        }
    }
}

extern "C" void kernel_launch(void* const* d_in, const int* in_sizes, int n_in,
                              void* d_out, int out_size, void* d_ws, size_t ws_size,
                              hipStream_t stream) {
    const float* inp  = (const float*)d_in[0];
    const int*   gate = (const int*)d_in[1];
    const float* w1   = (const float*)d_in[2];   // [E, HID, IN]
    const float* w2   = (const float*)d_in[3];   // [E, OUT, HID]
    float*       out  = (float*)d_out;           // [T, OUT]

    char* ws   = (char*)d_ws;
    int*  cnt  = (int*)ws;                       // 16 ints
    int*  perm = (int*)(ws + 128);               // 4096 ints
    float* h   = (float*)(ws + 65536);           // [T, HID] fp32 = 64 MB

    zero_counts_kernel<<<1, 32, 0, stream>>>(cnt);
    build_perm_kernel<<<T_TOKENS / 256, 256, 0, stream>>>(gate, cnt, perm);

    // layer 1: h[e,c,:] = W1[e] @ x_gathered   (A gathered via perm)
    moe_gemm_kernel<true, false>
        <<<dim3(HID_FEAT / BN, CAP / BM, NUM_EXPERT), 256, 0, stream>>>(
            inp, w1, h, perm, HID_FEAT, IN_FEAT);

    // layer 2: y[t,:] = W2[e] @ h[e,c,:]       (C scattered via perm)
    moe_gemm_kernel<false, true>
        <<<dim3(OUT_FEAT / BN, CAP / BM, NUM_EXPERT), 256, 0, stream>>>(
            h, w2, out, perm, OUT_FEAT, HID_FEAT);
}
